// SurfVQMAE_7017976562325
// MI455X (gfx1250) — compile-verified, matrix operand order checked by reading, not run on hardware
//
#include <hip/hip_runtime.h>

// ---------------------------------------------------------------------------
// CDNA5 (gfx1250) implementation of SurfVQMAE forward.
// All heavy GEMMs use v_wmma_f32_16x16x32_bf16 (bf16 inputs, f32 accumulate).
// Attention is flash-style with WMMA tiles (hd=32 == WMMA K).
// Data movement: 128-bit global loads -> LDS bf16 tiles -> ds_load_b128 frags.
// ---------------------------------------------------------------------------

typedef __bf16 bf16;
typedef __attribute__((ext_vector_type(16))) __bf16 v16bf;
typedef __attribute__((ext_vector_type(8)))  float  v8f;

union AFrag { v16bf v; bf16  e[16]; };
union CFrag { v8f   v; float e[8];  };

__device__ __forceinline__ bf16 f2bf(float f) {
  unsigned u = __float_as_uint(f);
  unsigned r = u + 0x7FFFu + ((u >> 16) & 1u);   // round-to-nearest-even
  union { unsigned short s; bf16 b; } o; o.s = (unsigned short)(r >> 16);
  return o.b;
}
__device__ __forceinline__ float bf2f(bf16 b) {
  union { bf16 b; unsigned short s; } i; i.b = b;
  return __uint_as_float(((unsigned)i.s) << 16);
}
__device__ __forceinline__ bf16 bf_zero() {
  union { unsigned short s; bf16 b; } z; z.s = 0; return z.b;
}

// Convert 8 contiguous f32 -> 8 contiguous bf16 with one b128 store.
// src and dst must be 16B-aligned.
__device__ __forceinline__ void load8_f32_to_bf16(const float* __restrict__ src,
                                                  bf16* dst) {
  float4 a = *(const float4*)src;
  float4 b = *(const float4*)(src + 4);
  union { uint4 u; bf16 e[8]; } t;
  t.e[0] = f2bf(a.x); t.e[1] = f2bf(a.y); t.e[2] = f2bf(a.z); t.e[3] = f2bf(a.w);
  t.e[4] = f2bf(b.x); t.e[5] = f2bf(b.y); t.e[6] = f2bf(b.z); t.e[7] = f2bf(b.w);
  *(uint4*)dst = t.u;
}

// k-index of element i of a 16x32 (16-bit) A/B fragment held by `lane`
// (ISA 7.12.2): elements 0..7 are k = kb0..kb0+7, elements 8..15 are
// k = 16+kb0..16+kb0+7 with kb0 = (lane>>4)*8 -> two contiguous 16B groups.
__device__ __forceinline__ void frag_from_lds(AFrag* f, const bf16* row, int lane) {
  int kb0 = (lane >> 4) << 3;
  *(uint4*)&f->e[0] = *(const uint4*)(row + kb0);
  *(uint4*)&f->e[8] = *(const uint4*)(row + 16 + kb0);
}

__device__ __forceinline__ float block_sum(float v, float* sbuf) {
  int tid = threadIdx.x;
  sbuf[tid] = v; __syncthreads();
  for (int s = 128; s > 0; s >>= 1) {
    if (tid < s) sbuf[tid] += sbuf[tid + s];
    __syncthreads();
  }
  float r = sbuf[0]; __syncthreads();
  return r;
}

// ---------------------------------------------------------------------------
// Generic tiled WMMA GEMM: C[M,Nout] = act(A[M,K] @ Bw[K,Nstore] + bias) (+res)
// Block tile 64x64, 256 threads = 8 waves, each wave two 16x16 WMMA tiles.
// Preconditions: M%64==0, K%32==0, Nstore%64==0 (Bw zero-padded if needed).
// ABF: 0 = A is f32, 1 = A is bf16 (compile-time -> branch-free staging).
// act: 0=none 1=relu 2=gelu(exact). Cf (f32) and/or Cb (bf16) outputs.
// ---------------------------------------------------------------------------
template <int ABF>
__global__ __launch_bounds__(256) void gemm_wmma_kernel(
    const void* __restrict__ Aptr,
    const bf16* __restrict__ Bw,
    const float* __restrict__ bias,
    const float* __restrict__ res,
    float* __restrict__ Cf, bf16* __restrict__ Cb,
    int M, int Nstore, int K, int Nout, int act)
{
  __shared__ bf16 As[64][32];   // [m][k]
  __shared__ bf16 Bs[64][32];   // [n][k] (transposed stage -> contiguous frags)

  const int tid  = threadIdx.x;
  const int lane = tid & 31;
  const int wave = tid >> 5;
  const int wm   = wave >> 1;        // 0..3
  const int wn   = wave & 1;         // 0..1
  const int m0   = blockIdx.y * 64;
  const int n0   = blockIdx.x * 64;

  const float* Af = (const float*)Aptr;
  const bf16*  Ab = (const bf16*)Aptr;

  // staging coordinates (8 contiguous elements per thread, 128-bit accesses)
  const int arow = tid >> 2;            // 0..63
  const int akc  = (tid & 3) << 3;      // 0,8,16,24
  const int bnB  = (tid & 7) << 3;      // 0..56
  const int bkB  = tid >> 3;            // 0..31

  CFrag acc[2];
#pragma unroll
  for (int hh = 0; hh < 2; hh++)
#pragma unroll
    for (int r = 0; r < 8; r++) acc[hh].e[r] = 0.0f;

  for (int kk = 0; kk < K; kk += 32) {
    // ---- stage A (64x32): one b128 (or 2x float4 + cvt) per thread
    if (ABF) {
      *(uint4*)&As[arow][akc] =
          *(const uint4*)(Ab + (size_t)(m0 + arow) * K + kk + akc);
    } else {
      load8_f32_to_bf16(Af + (size_t)(m0 + arow) * K + kk + akc, &As[arow][akc]);
    }
    // ---- stage B (32x64 -> Bs[n][k]): one b128 global load per thread
    {
      union { uint4 u; bf16 e[8]; } t;
      t.u = *(const uint4*)(Bw + (size_t)(kk + bkB) * Nstore + n0 + bnB);
#pragma unroll
      for (int j = 0; j < 8; j++) Bs[bnB + j][bkB] = t.e[j];
    }
    __syncthreads();

    // prefetch next k-tile (CDNA5 global_prefetch_b8)
    if (kk + 32 < K) {
      if (ABF) __builtin_prefetch((const void*)(Ab + (size_t)(m0 + arow) * K + kk + 32 + akc), 0, 1);
      else     __builtin_prefetch((const void*)(Af + (size_t)(m0 + arow) * K + kk + 32 + akc), 0, 1);
      __builtin_prefetch((const void*)(Bw + (size_t)(kk + 32 + bkB) * Nstore + n0 + bnB), 0, 1);
    }

    // ---- compute: one A frag, two B frags, two wmma per wave per k-step
    AFrag a;
    frag_from_lds(&a, &As[wm * 16 + (lane & 15)][0], lane);
#pragma unroll
    for (int hh = 0; hh < 2; hh++) {
      AFrag bfr;
      frag_from_lds(&bfr, &Bs[wn * 32 + hh * 16 + (lane & 15)][0], lane);
      acc[hh].v = __builtin_amdgcn_wmma_f32_16x16x32_bf16(
          false, a.v, false, bfr.v, (short)0, acc[hh].v, false, false);
    }
    __syncthreads();
  }

  // ---- epilogue: C layout -> lanes 0-15 rows 0-7, lanes 16-31 rows 8-15
#pragma unroll
  for (int hh = 0; hh < 2; hh++)
#pragma unroll
    for (int r = 0; r < 8; r++) {
      int m = m0 + wm * 16 + ((lane >> 4) << 3) + r;
      int n = n0 + wn * 32 + hh * 16 + (lane & 15);
      if (m < M && n < Nout) {
        float v = acc[hh].e[r];
        if (bias) v += bias[n];
        if (act == 1)      v = fmaxf(v, 0.0f);
        else if (act == 2) v = 0.5f * v * (1.0f + erff(v * 0.70710678118654752f));
        size_t o = (size_t)m * Nout + n;
        if (res) v += res[o];
        if (Cf) Cf[o] = v;
        if (Cb) Cb[o] = f2bf(v);
      }
    }
}

// ---------------------------------------------------------------------------
// Flash attention, 1 wave per (b, h, 16-query tile). hd = 32 = WMMA K.
// Per 32-key chunk: stage K/Dm (coalesced float4) -> 2 wmma scores; inline
// RBF bias; online softmax via LDS; stage V transposed -> 2 wmma P@V.
// ---------------------------------------------------------------------------
__global__ __launch_bounds__(32) void attn_kernel(
    const float* __restrict__ qkv,   // (B*T, 3, H, 32) flattened to (B*T, 768)
    const float* __restrict__ Dm,    // (B, T, T)
    const float* __restrict__ rbf_w, // (16, H) slice for this layer
    const float* __restrict__ rbf_b, // (H,)
    float* __restrict__ out,         // (B*T, 256)
    int B, int T, int H, int hd)
{
  const int D  = H * hd;             // 256
  const int qt = blockIdx.x, h = blockIdx.y, b = blockIdx.z;
  const int q0 = qt * 16;
  const int lane = threadIdx.x;

  __shared__ bf16  Qld[16][32];      // [q][d]
  __shared__ bf16  Kld[32][32];      // [key][d]
  __shared__ bf16  Vld[32][32];      // [d][key]  (transposed for P@V frags)
  __shared__ float Dld[16][36];      // Dm tile (row stride 144B, 16B aligned)
  __shared__ float Sld[16][36];
  __shared__ bf16  Pld[16][32];
  __shared__ float rowM[16], rowS[16], rowC[16];

  float wb[16];
#pragma unroll
  for (int r = 0; r < 16; r++) wb[r] = rbf_w[r * H + h];
  const float bb    = rbf_b[h];
  const float beta  = 1.0f / (2.0f * 4.000001f * 4.000001f);
  const float scale = 0.17677669529663687f;          // 1/sqrt(32)

  // ---- stage Q (16 rows x 32 d): each lane converts half a row
  {
    int row = lane >> 1, c0 = (lane & 1) * 16;
    const float* src = qkv + ((size_t)(b * T + q0 + row) * 3 + 0) * D + h * hd + c0;
    load8_f32_to_bf16(src,     &Qld[row][c0]);
    load8_f32_to_bf16(src + 8, &Qld[row][c0 + 8]);
  }
  if (lane < 16) { rowM[lane] = -3.0e38f; rowS[lane] = 0.0f; }
  __syncthreads();

  AFrag qa;
  frag_from_lds(&qa, &Qld[lane & 15][0], lane);

  CFrag o0, o1;
#pragma unroll
  for (int r = 0; r < 8; r++) { o0.e[r] = 0.0f; o1.e[r] = 0.0f; }

  for (int kc = 0; kc < T; kc += 32) {
    // ---- stage K rows (each lane one 128B row) and Dm tile
    {
      const float* ks = qkv + ((size_t)(b * T + kc + lane) * 3 + 1) * D + h * hd;
      load8_f32_to_bf16(ks,      &Kld[lane][0]);
      load8_f32_to_bf16(ks + 8,  &Kld[lane][8]);
      load8_f32_to_bf16(ks + 16, &Kld[lane][16]);
      load8_f32_to_bf16(ks + 24, &Kld[lane][24]);
    }
    if (lane < 16) {
      const float* dsrc = Dm + ((size_t)(b * T + q0 + lane)) * T + kc;
#pragma unroll
      for (int g = 0; g < 8; g++)
        *(float4*)&Dld[lane][g * 4] = ((const float4*)dsrc)[g];
    }
    __syncthreads();

    // ---- scores: S = Q @ K^T (2 wmma)
    CFrag s[2];
#pragma unroll
    for (int hf = 0; hf < 2; hf++) {
      AFrag kb;
      frag_from_lds(&kb, &Kld[hf * 16 + (lane & 15)][0], lane);
      CFrag z;
#pragma unroll
      for (int r = 0; r < 8; r++) z.e[r] = 0.0f;
      s[hf].v = __builtin_amdgcn_wmma_f32_16x16x32_bf16(
          false, qa.v, false, kb.v, (short)0, z.v, false, false);
    }
    // ---- scale + inline RBF bias, spill to LDS
#pragma unroll
    for (int hf = 0; hf < 2; hf++)
#pragma unroll
      for (int r = 0; r < 8; r++) {
        int m = ((lane >> 4) << 3) + r;
        int n = hf * 16 + (lane & 15);
        float dg = Dld[m][n];
        float bias = bb;
#pragma unroll
        for (int j = 0; j < 16; j++) {
          float t = dg - 4.0f * (float)j;
          bias += wb[j] * __expf(-beta * t * t);
        }
        Sld[m][n] = s[hf].e[r] * scale + bias;
      }
    __syncthreads();

    // ---- online softmax (lanes 0..15 each own a query row)
    if (lane < 16) {
      int row = lane;
      float mx = rowM[row];
#pragma unroll
      for (int j = 0; j < 32; j++) mx = fmaxf(mx, Sld[row][j]);
      float c = __expf(rowM[row] - mx);
      float sum = 0.0f;
#pragma unroll
      for (int j = 0; j < 32; j++) {
        float p = __expf(Sld[row][j] - mx);
        Pld[row][j] = f2bf(p);
        sum += p;
      }
      rowS[row] = rowS[row] * c + sum;
      rowM[row] = mx;
      rowC[row] = c;
    }
    __syncthreads();

    // ---- stage V transposed: Vld[d][key]
    {
      const float* vs = qkv + ((size_t)(b * T + kc + lane) * 3 + 2) * D + h * hd;
#pragma unroll
      for (int g = 0; g < 8; g++) {
        float4 f = ((const float4*)vs)[g];
        Vld[g * 4 + 0][lane] = f2bf(f.x);
        Vld[g * 4 + 1][lane] = f2bf(f.y);
        Vld[g * 4 + 2][lane] = f2bf(f.z);
        Vld[g * 4 + 3][lane] = f2bf(f.w);
      }
    }
    // rescale accumulators by per-row correction
#pragma unroll
    for (int r = 0; r < 8; r++) {
      float c = rowC[((lane >> 4) << 3) + r];
      o0.e[r] *= c; o1.e[r] *= c;
    }
    __syncthreads();

    // ---- O += P @ V (2 wmma)
    AFrag pa;
    frag_from_lds(&pa, &Pld[lane & 15][0], lane);
#pragma unroll
    for (int hf = 0; hf < 2; hf++) {
      AFrag vb;
      frag_from_lds(&vb, &Vld[hf * 16 + (lane & 15)][0], lane);
      CFrag* op = hf ? &o1 : &o0;
      op->v = __builtin_amdgcn_wmma_f32_16x16x32_bf16(
          false, pa.v, false, vb.v, (short)0, op->v, false, false);
    }
    __syncthreads();
  }

  // ---- normalize + store (B,T,H,hd) packed as (B*T, 256)
#pragma unroll
  for (int r = 0; r < 8; r++) {
    int m = ((lane >> 4) << 3) + r;
    float inv = 1.0f / rowS[m];
    size_t base = ((size_t)(b * T + q0 + m)) * D + h * hd;
    out[base + (lane & 15)]      = o0.e[r] * inv;
    out[base + 16 + (lane & 15)] = o1.e[r] * inv;
  }
}

// ---------------------------------------------------------------------------
// Small helper kernels
// ---------------------------------------------------------------------------
__global__ void f32_to_bf16_kernel(const float* __restrict__ s, bf16* __restrict__ d, long long n) {
  long long i = (long long)blockIdx.x * blockDim.x + threadIdx.x;
  long long st = (long long)gridDim.x * blockDim.x;
  for (; i < n; i += st) d[i] = f2bf(s[i]);
}

__global__ void f32_to_bf16_pad_kernel(const float* __restrict__ s, bf16* __restrict__ d,
                                       int K, int Nreal, int Npad) {
  long long n = (long long)K * Npad;
  long long i = (long long)blockIdx.x * blockDim.x + threadIdx.x;
  long long st = (long long)gridDim.x * blockDim.x;
  for (; i < n; i += st) {
    int k = (int)(i / Npad), c = (int)(i % Npad);
    d[i] = (c < Nreal) ? f2bf(s[(size_t)k * Nreal + c]) : bf_zero();
  }
}

// first local layer: IN=6 -> 256, ReLU, bf16 out. One block per point row.
__global__ __launch_bounds__(256) void local_l1_kernel(
    const float* __restrict__ feats, const float* __restrict__ w1,
    const float* __restrict__ b1, bf16* __restrict__ h1) {
  __shared__ float f[6];
  int row = blockIdx.x, tid = threadIdx.x;
  if (tid < 6) f[tid] = feats[(size_t)row * 6 + tid];
  __syncthreads();
  float acc = b1[tid];
#pragma unroll
  for (int j = 0; j < 6; j++) acc += f[j] * w1[j * 256 + tid];
  h1[(size_t)row * 256 + tid] = f2bf(fmaxf(acc, 0.0f));
}

// max over K=50 points then LayerNorm -> x (f32). One block per (b,t).
__global__ __launch_bounds__(256) void maxln_kernel(
    const bf16* __restrict__ h3, const float* __restrict__ w,
    const float* __restrict__ b, float* __restrict__ x, int KP) {
  __shared__ float sbuf[256];
  int bt = blockIdx.x, d = threadIdx.x;
  float mx = -3.0e38f;
  for (int k = 0; k < KP; k++)
    mx = fmaxf(mx, bf2f(h3[((size_t)bt * KP + k) * 256 + d]));
  float mean = block_sum(mx, sbuf) * (1.0f / 256.0f);
  float df = mx - mean;
  float var = block_sum(df * df, sbuf) * (1.0f / 256.0f);
  x[(size_t)bt * 256 + d] = df * rsqrtf(var + 1e-5f) * w[d] + b[d];
}

// optional (+= positional encoding, in place) then LayerNorm -> outp.
__global__ __launch_bounds__(256) void ln_kernel(
    float* __restrict__ x, const float* __restrict__ w, const float* __restrict__ b,
    float* __restrict__ outp, int T, int addpe) {
  __shared__ float sbuf[256];
  int bt = blockIdx.x, d = threadIdx.x;
  float v = x[(size_t)bt * 256 + d];
  if (addpe) {
    int t = bt % T;
    float div = __expf(-(float)(d & ~1) * (9.210340371976184f / 256.0f));
    float ang = (float)t * div;
    v += (d & 1) ? __cosf(ang) : __sinf(ang);
    x[(size_t)bt * 256 + d] = v;
  }
  float mean = block_sum(v, sbuf) * (1.0f / 256.0f);
  float df = v - mean;
  float var = block_sum(df * df, sbuf) * (1.0f / 256.0f);
  outp[(size_t)bt * 256 + d] = df * rsqrtf(var + 1e-5f) * w[d] + b[d];
}

// pairwise distances Dm[b,q,k]. One block per (b,q).
__global__ __launch_bounds__(256) void dm_kernel(
    const float* __restrict__ centers, float* __restrict__ Dm, int T) {
  int bt = blockIdx.x, b = bt / T;
  float cx = centers[(size_t)bt * 3 + 0];
  float cy = centers[(size_t)bt * 3 + 1];
  float cz = centers[(size_t)bt * 3 + 2];
  for (int k = threadIdx.x; k < T; k += 256) {
    const float* ck = centers + ((size_t)b * T + k) * 3;
    float dx = cx - ck[0], dy = cy - ck[1], dz = cz - ck[2];
    Dm[(size_t)bt * T + k] = sqrtf(fmaxf(dx * dx + dy * dy + dz * dz, 0.0f));
  }
}

// VQ: argmax(logits+gumbel) -> one-hot y, tokens = mask ? codebook[idx] : xn
__global__ __launch_bounds__(256) void vq_kernel(
    const float* __restrict__ logits, const float* __restrict__ gumbel,
    const int* __restrict__ mask, const float* __restrict__ codebook,
    const float* __restrict__ xn, float* __restrict__ y_out,
    float* __restrict__ tokens, int NC) {
  __shared__ float bv[256];
  __shared__ int   bi[256];
  int bt = blockIdx.x, tid = threadIdx.x;
  float best = -3.4e38f; int bidx = 0;
  for (int j = tid; j < NC; j += 256) {
    float v = logits[(size_t)bt * NC + j] + gumbel[(size_t)bt * NC + j];
    if (v > best) { best = v; bidx = j; }
  }
  bv[tid] = best; bi[tid] = bidx; __syncthreads();
  for (int s = 128; s > 0; s >>= 1) {
    if (tid < s && bv[tid + s] > bv[tid]) { bv[tid] = bv[tid + s]; bi[tid] = bi[tid + s]; }
    __syncthreads();
  }
  int amax = bi[0];
  for (int j = tid; j < NC; j += 256)
    y_out[(size_t)bt * NC + j] = (j == amax) ? 1.0f : 0.0f;
  int m = mask[bt];
  tokens[(size_t)bt * 256 + tid] =
      m ? codebook[(size_t)amax * 256 + tid] : xn[(size_t)bt * 256 + tid];
}

// curvature head final dot: curv[bt] = sum_d cbuf[bt,d]*w2[d] + b2
__global__ __launch_bounds__(256) void curv2_kernel(
    const float* __restrict__ cbuf, const float* __restrict__ w2,
    const float* __restrict__ b2, float* __restrict__ outc) {
  __shared__ float sbuf[256];
  int bt = blockIdx.x, tid = threadIdx.x;
  float p = cbuf[(size_t)bt * 256 + tid] * w2[tid];
  float s = block_sum(p, sbuf);
  if (tid == 0) outc[bt] = s + b2[0];
}

// ---------------------------------------------------------------------------
// Host-side orchestration
// ---------------------------------------------------------------------------
static inline void launch_gemm(const void* A, int abf, const bf16* Bw,
                               const float* bias, const float* res,
                               float* Cf, bf16* Cb,
                               int M, int Nstore, int K, int Nout, int act,
                               hipStream_t s) {
  dim3 g(Nstore / 64, M / 64);
  if (abf)
    gemm_wmma_kernel<1><<<g, 256, 0, s>>>(A, Bw, bias, res, Cf, Cb, M, Nstore, K, Nout, act);
  else
    gemm_wmma_kernel<0><<<g, 256, 0, s>>>(A, Bw, bias, res, Cf, Cb, M, Nstore, K, Nout, act);
}

static inline void launch_conv(const float* s, bf16* d, long long n, hipStream_t st) {
  int blocks = (int)((n + 255) / 256); if (blocks > 2048) blocks = 2048;
  f32_to_bf16_kernel<<<blocks, 256, 0, st>>>(s, d, n);
}

extern "C" void kernel_launch(void* const* d_in, const int* in_sizes, int n_in,
                              void* d_out, int out_size, void* d_ws, size_t ws_size,
                              hipStream_t stream) {
  (void)in_sizes; (void)n_in; (void)out_size; (void)ws_size;
  const int B = 4, T = 1024, KP = 50, D = 256, H = 8, HD = 32, L = 6;
  const int NC = 2048, FF = 1024, DH = 512;
  const int BT = B * T;            // 4096
  const int MPT = B * T * KP;      // 204800

  const float* feats     = (const float*)d_in[0];
  const float* centers   = (const float*)d_in[1];
  const int*   mask      = (const int*)  d_in[2];
  const float* gumbel    = (const float*)d_in[3];
  const float* local_w1  = (const float*)d_in[4];
  const float* local_b1  = (const float*)d_in[5];
  const float* local_w2  = (const float*)d_in[6];
  const float* local_b2  = (const float*)d_in[7];
  const float* local_w3  = (const float*)d_in[8];
  const float* local_b3  = (const float*)d_in[9];
  const float* local_lnw = (const float*)d_in[10];
  const float* local_lnb = (const float*)d_in[11];
  const float* qkv_w     = (const float*)d_in[12];
  const float* proj_w    = (const float*)d_in[13];
  const float* proj_b    = (const float*)d_in[14];
  const float* ln1_w     = (const float*)d_in[15];
  const float* ln1_b     = (const float*)d_in[16];
  const float* ln2_w     = (const float*)d_in[17];
  const float* ln2_b     = (const float*)d_in[18];
  const float* ff1_w     = (const float*)d_in[19];
  const float* ff1_b     = (const float*)d_in[20];
  const float* ff2_w     = (const float*)d_in[21];
  const float* ff2_b     = (const float*)d_in[22];
  const float* rbf_w     = (const float*)d_in[23];
  const float* rbf_b     = (const float*)d_in[24];
  const float* pre_lnw   = (const float*)d_in[25];
  const float* pre_lnb   = (const float*)d_in[26];
  const float* logits_w  = (const float*)d_in[27];
  const float* logits_b  = (const float*)d_in[28];
  const float* codebook  = (const float*)d_in[29];
  const float* dec_w1    = (const float*)d_in[30];
  const float* dec_b1    = (const float*)d_in[31];
  const float* dec_w2    = (const float*)d_in[32];
  const float* dec_b2    = (const float*)d_in[33];
  const float* dec_w3    = (const float*)d_in[34];
  const float* dec_b3    = (const float*)d_in[35];
  const float* curv_lnw  = (const float*)d_in[36];
  const float* curv_lnb  = (const float*)d_in[37];
  const float* curv_w1   = (const float*)d_in[38];
  const float* curv_b1   = (const float*)d_in[39];
  const float* curv_w2   = (const float*)d_in[40];
  const float* curv_b2   = (const float*)d_in[41];

  // ---- output regions (return order: rec, curv, logits, y)
  float* out_rec    = (float*)d_out;
  float* out_curv   = out_rec + (size_t)BT * KP * 3;     // 614400
  float* out_logits = out_curv + BT;                     // +4096
  float* out_y      = out_logits + (size_t)BT * NC;      // +8388608

  // ---- workspace carve
  char* cur = (char*)d_ws;
  auto alloc = [&](size_t bytes) -> char* {
    char* p = cur; cur += (bytes + 255) & ~(size_t)255; return p;
  };
  float* Dm       = (float*)alloc((size_t)B * T * T * 4);       // 16.8 MB
  bf16*  h1       = (bf16*) alloc((size_t)MPT * D * 2);         // 104.9 MB
  bf16*  h2       = (bf16*) alloc((size_t)MPT * D * 2);         // 104.9 MB
  float* x        = (float*)alloc((size_t)BT * D * 4);
  float* hN       = (float*)alloc((size_t)BT * D * 4);
  float* qkvb     = (float*)alloc((size_t)BT * 3 * D * 4);      // 12.6 MB
  float* attn_out = (float*)alloc((size_t)BT * D * 4);
  float* ffb      = (float*)alloc((size_t)BT * FF * 4);         // 16.8 MB
  float* tokens   = (float*)alloc((size_t)BT * D * 4);
  bf16*  w2bf     = (bf16*) alloc((size_t)256 * 256 * 2);
  bf16*  w3bf     = (bf16*) alloc((size_t)256 * 256 * 2);
  bf16*  qkvbf    = (bf16*) alloc((size_t)L * 256 * 768 * 2);
  bf16*  projbf   = (bf16*) alloc((size_t)L * 256 * 256 * 2);
  bf16*  ff1bf    = (bf16*) alloc((size_t)L * 256 * FF * 2);
  bf16*  ff2bf    = (bf16*) alloc((size_t)L * FF * 256 * 2);
  bf16*  logitsbf = (bf16*) alloc((size_t)256 * NC * 2);
  bf16*  dw1bf    = (bf16*) alloc((size_t)256 * DH * 2);
  bf16*  dw2bf    = (bf16*) alloc((size_t)DH * DH * 2);
  bf16*  dw3bf    = (bf16*) alloc((size_t)DH * 192 * 2);        // padded 150->192
  bf16*  cw1bf    = (bf16*) alloc((size_t)256 * 256 * 2);
  bf16*  h3   = h1;        // aliases (lifetimes disjoint)
  float* d1b  = ffb;
  float* d2b  = qkvb;
  float* ctb  = hN;
  float* cbuf = attn_out;

  // ---- weight conversion to bf16 (deterministic, every call)
  launch_conv(local_w2, w2bf,     256 * 256,            stream);
  launch_conv(local_w3, w3bf,     256 * 256,            stream);
  launch_conv(qkv_w,    qkvbf,    (long long)L * 256 * 768, stream);
  launch_conv(proj_w,   projbf,   (long long)L * 256 * 256, stream);
  launch_conv(ff1_w,    ff1bf,    (long long)L * 256 * FF,  stream);
  launch_conv(ff2_w,    ff2bf,    (long long)L * FF * 256,  stream);
  launch_conv(logits_w, logitsbf, 256 * NC,             stream);
  launch_conv(dec_w1,   dw1bf,    256 * DH,             stream);
  launch_conv(dec_w2,   dw2bf,    DH * DH,              stream);
  f32_to_bf16_pad_kernel<<<512, 256, 0, stream>>>(dec_w3, dw3bf, DH, KP * 3, 192);
  launch_conv(curv_w1,  cw1bf,    256 * 256,            stream);

  // ---- geometry distances
  dm_kernel<<<BT, 256, 0, stream>>>(centers, Dm, T);

  // ---- local per-point MLP (dominant GFLOPs -> WMMA)
  local_l1_kernel<<<MPT, 256, 0, stream>>>(feats, local_w1, local_b1, h1);
  launch_gemm(h1, 1, w2bf, local_b2, nullptr, nullptr, h2, MPT, 256, 256, 256, 1, stream);
  launch_gemm(h2, 1, w3bf, local_b3, nullptr, nullptr, h3, MPT, 256, 256, 256, 0, stream);
  maxln_kernel<<<BT, 256, 0, stream>>>(h3, local_lnw, local_lnb, x, KP);

  // ---- transformer stack
  for (int l = 0; l < L; l++) {
    ln_kernel<<<BT, 256, 0, stream>>>(x, ln1_w + l * 256, ln1_b + l * 256, hN, T, 1);
    launch_gemm(hN, 0, qkvbf + (size_t)l * 256 * 768, nullptr, nullptr,
                qkvb, nullptr, BT, 768, 256, 768, 0, stream);
    attn_kernel<<<dim3(T / 16, H, B), 32, 0, stream>>>(
        qkvb, Dm, rbf_w + l * 16 * H, rbf_b + l * H, attn_out, B, T, H, HD);
    launch_gemm(attn_out, 0, projbf + (size_t)l * 256 * 256, proj_b + l * 256, x,
                x, nullptr, BT, 256, 256, 256, 0, stream);
    ln_kernel<<<BT, 256, 0, stream>>>(x, ln2_w + l * 256, ln2_b + l * 256, hN, T, 0);
    launch_gemm(hN, 0, ff1bf + (size_t)l * 256 * FF, ff1_b + l * FF, nullptr,
                ffb, nullptr, BT, FF, 256, FF, 2, stream);
    launch_gemm(ffb, 0, ff2bf + (size_t)l * FF * 256, ff2_b + l * 256, x,
                x, nullptr, BT, 256, FF, 256, 0, stream);
  }

  // ---- head: pre-LN, logits (straight to d_out), VQ straight-through
  ln_kernel<<<BT, 256, 0, stream>>>(x, pre_lnw, pre_lnb, hN, T, 0);
  launch_gemm(hN, 0, logitsbf, logits_b, nullptr, out_logits, nullptr,
              BT, NC, 256, NC, 0, stream);
  vq_kernel<<<BT, 256, 0, stream>>>(out_logits, gumbel, mask, codebook, hN,
                                    out_y, tokens, NC);

  // ---- decoder -> rec
  launch_gemm(tokens, 0, dw1bf, dec_b1, nullptr, d1b, nullptr, BT, DH, 256, DH, 2, stream);
  launch_gemm(d1b,    0, dw2bf, dec_b2, nullptr, d2b, nullptr, BT, DH, DH, DH, 2, stream);
  launch_gemm(d2b,    0, dw3bf, dec_b3, nullptr, out_rec, nullptr,
              BT, 192, DH, KP * 3, 0, stream);

  // ---- curvature head -> curv
  ln_kernel<<<BT, 256, 0, stream>>>(tokens, curv_lnw, curv_lnb, ctb, T, 0);
  launch_gemm(ctb, 0, cw1bf, curv_b1, nullptr, cbuf, nullptr, BT, 256, 256, 256, 2, stream);
  curv2_kernel<<<BT, 256, 0, stream>>>(cbuf, curv_w2, curv_b2, out_curv);
}
